// spe_dense_bottleneck_1D_76398878261342
// MI455X (gfx1250) — matrix operand order repaired and checked by reading, not compiled, not run on hardware
//
#include <hip/hip_runtime.h>
#include <hip/hip_bf16.h>

typedef __attribute__((ext_vector_type(16))) __bf16 v16bf;
typedef __attribute__((ext_vector_type(8)))  float  v8f;

struct FragU { unsigned u[8]; };

__device__ __forceinline__ unsigned short f2bf(float f) {
  unsigned u = __builtin_bit_cast(unsigned, f);
  return (unsigned short)((u + 0x7FFFu + ((u >> 16) & 1u)) >> 16);
}
__device__ __forceinline__ float bf2f(unsigned short u) {
  return __builtin_bit_cast(float, ((unsigned)u) << 16);
}
__device__ __forceinline__ unsigned pack_bf2(float lo, float hi) {
  return (unsigned)f2bf(lo) | ((unsigned)f2bf(hi) << 16);
}

// A fragment (16x32 bf16, MxK), source row-major [rows][ld] bf16.
// lane m=lane&15, hi=lane>>4: V0..3 hold K = 8*hi + {0,2,4,6}(+1); V4..7 hold K = 16+8*hi + {0,2,4,6}(+1)
__device__ __forceinline__ v16bf load_a_frag(const unsigned short* A, int ld, int mrow, int kb, int lane) {
  int m = lane & 15, hi = (lane >> 4) & 1;
  const unsigned short* p = A + (size_t)(mrow + m) * ld + kb + 8 * hi;
  FragU f;
#pragma unroll
  for (int v = 0; v < 4; ++v) f.u[v]     = *(const unsigned*)(p + 2 * v);
#pragma unroll
  for (int v = 0; v < 4; ++v) f.u[4 + v] = *(const unsigned*)(p + 16 + 2 * v);
  return __builtin_bit_cast(v16bf, f);
}

// B fragment (32x16 bf16, KxN), source stored N-major: Bt[n][ld] holds K along rows.
// lane n=lane&15, hi=lane>>4: V0..3 hold K = 8*hi + {0,2,4,6}(+1); V4..7 hold K = 16+8*hi + {0,2,4,6}(+1), column n.
__device__ __forceinline__ v16bf load_b_frag(const unsigned short* Bt, int ld, int ncol, int kb, int lane) {
  int n = lane & 15, hi = (lane >> 4) & 1;
  const unsigned short* p = Bt + (size_t)(ncol + n) * ld + kb + 8 * hi;
  FragU f;
#pragma unroll
  for (int v = 0; v < 8; ++v) f.u[v] = *(const unsigned*)(p + 2 * (v & 3) + 16 * (v >> 2));
  return __builtin_bit_cast(v16bf, f);
}

// Same fragments but built from fp32 source (converted to bf16 on the fly)
__device__ __forceinline__ v16bf load_a_frag_f32(const float* A, int ld, int mrow, int kb, int lane) {
  int m = lane & 15, hi = (lane >> 4) & 1;
  const float* p = A + (size_t)(mrow + m) * ld + kb + 8 * hi;
  FragU f;
#pragma unroll
  for (int v = 0; v < 4; ++v) f.u[v]     = pack_bf2(p[2 * v], p[2 * v + 1]);
#pragma unroll
  for (int v = 0; v < 4; ++v) f.u[4 + v] = pack_bf2(p[16 + 2 * v], p[17 + 2 * v]);
  return __builtin_bit_cast(v16bf, f);
}
__device__ __forceinline__ v16bf load_b_frag_f32(const float* H, int ld, int ncol, int kb, int lane) {
  int n = lane & 15, hi = (lane >> 4) & 1;
  const float* p = H + (size_t)(ncol + n) * ld + kb + 8 * hi;
  FragU f;
#pragma unroll
  for (int v = 0; v < 8; ++v) f.u[v] = pack_bf2(p[2 * (v & 3) + 16 * (v >> 2)], p[2 * (v & 3) + 16 * (v >> 2) + 1]);
  return __builtin_bit_cast(v16bf, f);
}

#define WMMA_BF16(a, b, c) \
  __builtin_amdgcn_wmma_f32_16x16x32_bf16(false, (a), false, (b), (short)0, (c), false, false)

// ---------------- BN stats (deterministic two-stage) ----------------
__global__ void __launch_bounds__(256) k_part_f32(const float* __restrict__ x,
                                                  float* __restrict__ ps, float* __restrict__ pq,
                                                  int C, int L, int B) {
  int bc = blockIdx.x;
  int b = bc / C, c = bc - b * C;
  const float* row = x + (size_t)bc * L;
  float s = 0.f, q = 0.f;
  for (int i = threadIdx.x; i < L; i += 256) { float v = row[i]; s += v; q += v * v; }
  __shared__ float rs[256], rq[256];
  rs[threadIdx.x] = s; rq[threadIdx.x] = q; __syncthreads();
  for (int o = 128; o > 0; o >>= 1) {
    if (threadIdx.x < o) { rs[threadIdx.x] += rs[threadIdx.x + o]; rq[threadIdx.x] += rq[threadIdx.x + o]; }
    __syncthreads();
  }
  if (threadIdx.x == 0) { ps[(size_t)c * B + b] = rs[0]; pq[(size_t)c * B + b] = rq[0]; }
}

__global__ void __launch_bounds__(256) k_part_bf16(const unsigned short* __restrict__ h,
                                                   float* __restrict__ ps, float* __restrict__ pq,
                                                   int C, int L, int B) {
  int bc = blockIdx.x;
  int b = bc / C, c = bc - b * C;
  const unsigned short* row = h + (size_t)bc * L;
  float s = 0.f, q = 0.f;
  for (int i = threadIdx.x; i < L; i += 256) { float v = bf2f(row[i]); s += v; q += v * v; }
  __shared__ float rs[256], rq[256];
  rs[threadIdx.x] = s; rq[threadIdx.x] = q; __syncthreads();
  for (int o = 128; o > 0; o >>= 1) {
    if (threadIdx.x < o) { rs[threadIdx.x] += rs[threadIdx.x + o]; rq[threadIdx.x] += rq[threadIdx.x + o]; }
    __syncthreads();
  }
  if (threadIdx.x == 0) { ps[(size_t)c * B + b] = rs[0]; pq[(size_t)c * B + b] = rq[0]; }
}

__global__ void __launch_bounds__(256) k_reduce_stats(const float* __restrict__ ps, const float* __restrict__ pq,
                                                      float* __restrict__ sum, float* __restrict__ sq, int B) {
  int c = blockIdx.x, t = threadIdx.x;
  __shared__ float rs[256], rq[256];
  float s = (t < B) ? ps[(size_t)c * B + t] : 0.f;
  float q = (t < B) ? pq[(size_t)c * B + t] : 0.f;
  rs[t] = s; rq[t] = q; __syncthreads();
  for (int o = 128; o > 0; o >>= 1) {
    if (t < o) { rs[t] += rs[t + o]; rq[t] += rq[t + o]; }
    __syncthreads();
  }
  if (t == 0) { sum[c] = rs[0]; sq[c] = rq[0]; }
}

__global__ void k_finalize(const float* __restrict__ sum, const float* __restrict__ sq,
                           const float* __restrict__ gamma, const float* __restrict__ beta,
                           float* __restrict__ scale, float* __restrict__ shift, int C, float invN) {
  int c = blockIdx.x * blockDim.x + threadIdx.x;
  if (c >= C) return;
  float m = sum[c] * invN;
  float v = sq[c] * invN - m * m;
  float sc = gamma[c] * rsqrtf(v + 1e-5f);
  scale[c] = sc;
  shift[c] = beta[c] - m * sc;
}

// ---------------- weight conversions ----------------
__global__ void k_cvt_w1(const float* __restrict__ W1, unsigned short* __restrict__ w1bf, int n) {
  int i = blockIdx.x * blockDim.x + threadIdx.x;
  if (i < n) w1bf[i] = f2bf(W1[i]);
}
// W2 [48,192,3] fp32 -> three row-major A matrices A[k][48][192] bf16
__global__ void k_cvt_w2(const float* __restrict__ W2, unsigned short* __restrict__ w2bf) {
  int i = blockIdx.x * blockDim.x + threadIdx.x;
  if (i >= 3 * 48 * 192) return;
  int k = i / (48 * 192);
  int r = i - k * (48 * 192);
  int co = r / 192, ci = r - co * 192;
  w2bf[i] = f2bf(W2[(co * 192 + ci) * 3 + k]);
}

// ---------------- conv1: h1[b] (192x2048 bf16) = W1(192x64) @ relu(bn1(x_b))(64x2048) ----------------
__global__ void __launch_bounds__(256) k_conv1(const float* __restrict__ x,
                                               const float* __restrict__ scale1, const float* __restrict__ shift1,
                                               const unsigned short* __restrict__ w1bf,
                                               unsigned short* __restrict__ h1) {
  __shared__ __align__(16) unsigned short Bl[128 * 64];  // yT[l][ci]
  __shared__ float s1[64], t1[64];
  const int tid = threadIdx.x;
  const int b = blockIdx.z, M0 = blockIdx.y * 64, L0 = blockIdx.x * 128;
  if (tid < 64) { s1[tid] = scale1[tid]; t1[tid] = shift1[tid]; }
  __syncthreads();
  const float* xb = x + (size_t)b * 64 * 2048;
  for (int i = tid; i < 64 * 128; i += 256) {
    int ci = i >> 7, l = i & 127;
    float v = xb[(size_t)ci * 2048 + L0 + l];
    float y = fmaxf(v * s1[ci] + t1[ci], 0.f);
    Bl[l * 64 + ci] = f2bf(y);
  }
  __syncthreads();
  const int lane = tid & 31, wave = tid >> 5;
  const int n0 = wave * 16;
  v8f acc[4] = {};
#pragma unroll
  for (int kb = 0; kb < 64; kb += 32) {
    v16bf bf = load_b_frag(Bl, 64, n0, kb, lane);
#pragma unroll
    for (int mt = 0; mt < 4; ++mt) {
      v16bf af = load_a_frag(w1bf + (size_t)M0 * 64, 64, mt * 16, kb, lane);
      acc[mt] = WMMA_BF16(af, bf, acc[mt]);
    }
  }
  const int hi = (lane >> 4) & 1, nl = lane & 15;
#pragma unroll
  for (int mt = 0; mt < 4; ++mt)
#pragma unroll
    for (int r = 0; r < 8; ++r) {
      int m = M0 + mt * 16 + r + 8 * hi;
      h1[((size_t)b * 192 + m) * 2048 + L0 + n0 + nl] = f2bf(acc[mt][r]);
    }
}

// ---------------- conv2 (k=3, pad=1): out5pre[b] (48x2048 f32, stored in d_out ch 64..111) ----------------
__global__ void __launch_bounds__(256) k_conv2(const unsigned short* __restrict__ h1,
                                               const float* __restrict__ scale2, const float* __restrict__ shift2,
                                               const unsigned short* __restrict__ w2bf,
                                               float* __restrict__ out) {
  __shared__ __align__(16) unsigned short Zl[130 * 192];  // zT[l_local][ci], l_local 0..129 = l in [L0-1, L0+128]
  __shared__ float s2[192], t2[192];
  const int tid = threadIdx.x;
  const int b = blockIdx.z, L0 = blockIdx.x * 128;
  if (tid < 192) { s2[tid] = scale2[tid]; t2[tid] = shift2[tid]; }
  __syncthreads();
  const unsigned short* hb = h1 + (size_t)b * 192 * 2048;
  for (int i = tid; i < 192 * 130; i += 256) {
    int ci = i / 130, ll = i - ci * 130;
    int gl = L0 - 1 + ll;
    float v = 0.f;
    if ((unsigned)gl < 2048u) v = bf2f(hb[(size_t)ci * 2048 + gl]);
    float z = fmaxf(v * s2[ci] + t2[ci], 0.f);
    Zl[ll * 192 + ci] = f2bf(z);
  }
  __syncthreads();
  const int lane = tid & 31, wave = tid >> 5;
  const int n0 = wave * 16;
  v8f acc[3] = {};
#pragma unroll
  for (int s = 0; s < 3; ++s) {
    const unsigned short* A = w2bf + (size_t)s * 48 * 192;
#pragma unroll
    for (int kb = 0; kb < 192; kb += 32) {
      v16bf bf = load_b_frag(Zl, 192, n0 + s, kb, lane);
#pragma unroll
      for (int mt = 0; mt < 3; ++mt) {
        v16bf af = load_a_frag(A, 192, mt * 16, kb, lane);
        acc[mt] = WMMA_BF16(af, bf, acc[mt]);
      }
    }
  }
  const int hi = (lane >> 4) & 1, nl = lane & 15;
  float* ob = out + ((size_t)b * 112 + 64) * 2048;
#pragma unroll
  for (int mt = 0; mt < 3; ++mt)
#pragma unroll
    for (int r = 0; r < 8; ++r) {
      int m = mt * 16 + r + 8 * hi;
      ob[(size_t)m * 2048 + L0 + n0 + nl] = acc[mt][r];
    }
}

// ---------------- Gram + softmax: att[b] (48x64 bf16, K-padded with zeros) ----------------
__global__ void __launch_bounds__(256) k_gram(const float* __restrict__ out, unsigned short* __restrict__ attw) {
  __shared__ float Gp[4][48 * 48];   // 36 KB
  __shared__ float G[48 * 48];       // 9 KB
  __shared__ float nrm[48];
  const int tid = threadIdx.x, lane = tid & 31, wave = tid >> 5;
  const int b = blockIdx.x;
  const float* h = out + ((size_t)b * 112 + 64) * 2048;   // 48 x 2048
  v8f acc[3][3] = {};
  const int kbase = wave * 256;
  for (int kk = 0; kk < 256; kk += 32) {
    int kb = kbase + kk;
    v16bf af[3], bf[3];
#pragma unroll
    for (int t = 0; t < 3; ++t) af[t] = load_a_frag_f32(h, 2048, t * 16, kb, lane);
#pragma unroll
    for (int t = 0; t < 3; ++t) bf[t] = load_b_frag_f32(h, 2048, t * 16, kb, lane);
#pragma unroll
    for (int mt = 0; mt < 3; ++mt)
#pragma unroll
      for (int nt = 0; nt < 3; ++nt)
        acc[mt][nt] = WMMA_BF16(af[mt], bf[nt], acc[mt][nt]);
  }
  const int hi = (lane >> 4) & 1, nl = lane & 15;
  // deterministic two-phase reduction across 8 waves (4 at a time; fixed order)
  if (wave < 4) {
#pragma unroll
    for (int mt = 0; mt < 3; ++mt)
#pragma unroll
      for (int nt = 0; nt < 3; ++nt)
#pragma unroll
        for (int r = 0; r < 8; ++r)
          Gp[wave][(mt * 16 + r + 8 * hi) * 48 + nt * 16 + nl] = acc[mt][nt][r];
  }
  __syncthreads();
  for (int i = tid; i < 48 * 48; i += 256) G[i] = Gp[0][i] + Gp[1][i] + Gp[2][i] + Gp[3][i];
  __syncthreads();
  if (wave >= 4) {
#pragma unroll
    for (int mt = 0; mt < 3; ++mt)
#pragma unroll
      for (int nt = 0; nt < 3; ++nt)
#pragma unroll
        for (int r = 0; r < 8; ++r)
          Gp[wave - 4][(mt * 16 + r + 8 * hi) * 48 + nt * 16 + nl] = acc[mt][nt][r];
  }
  __syncthreads();
  for (int i = tid; i < 48 * 48; i += 256) G[i] += Gp[0][i] + Gp[1][i] + Gp[2][i] + Gp[3][i];
  __syncthreads();
  if (tid < 48) nrm[tid] = sqrtf(fmaxf(G[tid * 48 + tid], 0.f));
  __syncthreads();
  if (tid < 48) {
    float nm = nrm[tid];
    float mx = -1e30f;
    for (int n = 0; n < 48; ++n) {
      float s = G[tid * 48 + n] / fmaxf(nm * nrm[n], 1e-8f);
      mx = fmaxf(mx, s);
    }
    float sum = 0.f;
    for (int n = 0; n < 48; ++n) {
      float s = G[tid * 48 + n] / fmaxf(nm * nrm[n], 1e-8f);
      float e = __expf(s - mx);
      G[tid * 48 + n] = e;   // only this thread touches this row
      sum += e;
    }
    float inv = 1.f / sum;
    unsigned short* ab = attw + (size_t)b * 48 * 64 + tid * 64;
    for (int n = 0; n < 48; ++n) ab[n] = f2bf(G[tid * 48 + n] * inv);
    for (int n = 48; n < 64; ++n) ab[n] = 0;
  }
}

// ---------------- att @ h + h, written over the h2 staging region of d_out ----------------
__global__ void __launch_bounds__(256) k_attn(const unsigned short* __restrict__ attw, float* __restrict__ out) {
  __shared__ __align__(16) unsigned short Hl[256 * 64];  // hT[l][k], K padded to 64
  const int tid = threadIdx.x, lane = tid & 31, wave = tid >> 5;
  const int b = blockIdx.z, L0 = blockIdx.x * 256;
  float* h = out + ((size_t)b * 112 + 64) * 2048;
  for (int i = tid; i < 48 * 256; i += 256) {
    int ci = i >> 8, l = i & 255;
    Hl[l * 64 + ci] = f2bf(h[(size_t)ci * 2048 + L0 + l]);
  }
  for (int i = tid; i < 16 * 256; i += 256) {
    int l = i >> 4, j = i & 15;
    Hl[l * 64 + 48 + j] = 0;
  }
  __syncthreads();
  const unsigned short* ab = attw + (size_t)b * 48 * 64;
  const int n0 = wave * 32;
  v8f acc[3][2] = {};
#pragma unroll
  for (int kb = 0; kb < 64; kb += 32) {
    v16bf bf[2];
#pragma unroll
    for (int nt = 0; nt < 2; ++nt) bf[nt] = load_b_frag(Hl, 64, n0 + nt * 16, kb, lane);
#pragma unroll
    for (int mt = 0; mt < 3; ++mt) {
      v16bf af = load_a_frag(ab, 64, mt * 16, kb, lane);
#pragma unroll
      for (int nt = 0; nt < 2; ++nt) acc[mt][nt] = WMMA_BF16(af, bf[nt], acc[mt][nt]);
    }
  }
  const int hi = (lane >> 4) & 1, nl = lane & 15;
#pragma unroll
  for (int mt = 0; mt < 3; ++mt)
#pragma unroll
    for (int nt = 0; nt < 2; ++nt)
#pragma unroll
      for (int r = 0; r < 8; ++r) {
        int m = mt * 16 + r + 8 * hi;
        size_t off = (size_t)m * 2048 + L0 + n0 + nt * 16 + nl;
        h[off] = acc[mt][nt][r] + h[off];   // residual; this WG owns this tile exclusively
      }
}

// ---------------- concat: x -> out channels 0..63 ----------------
__global__ void __launch_bounds__(256) k_copyx(const float* __restrict__ x, float* __restrict__ out) {
  const size_t N = (size_t)256 * 64 * 2048;
  const size_t stride = (size_t)gridDim.x * blockDim.x;
  for (size_t i = (size_t)blockIdx.x * blockDim.x + threadIdx.x; i < N; i += stride) {
    size_t b = i / (64 * 2048);
    size_t r = i - b * (64 * 2048);
    out[b * (112 * 2048) + r] = x[i];
  }
}

extern "C" void kernel_launch(void* const* d_in, const int* in_sizes, int n_in,
                              void* d_out, int out_size, void* d_ws, size_t ws_size,
                              hipStream_t stream) {
  const float* x      = (const float*)d_in[0];
  const float* gamma1 = (const float*)d_in[1];
  const float* beta1  = (const float*)d_in[2];
  const float* W1     = (const float*)d_in[3];
  const float* gamma2 = (const float*)d_in[4];
  const float* beta2  = (const float*)d_in[5];
  const float* W2     = (const float*)d_in[6];
  float* out = (float*)d_out;
  char* ws = (char*)d_ws;

  // workspace layout (bytes)
  float* sum1 = (float*)ws;            // 64
  float* sq1  = sum1 + 64;             // 64
  float* sum2 = sum1 + 128;            // 192
  float* sq2  = sum1 + 320;            // 192  (ends at 2048 B)
  float* scale1 = (float*)(ws + 2048); // 64
  float* shift1 = scale1 + 64;
  float* scale2 = scale1 + 128;
  float* shift2 = scale1 + 320;        // ends 4096 B
  float* ps1 = (float*)(ws + 4096);            // 64*256
  float* pq1 = ps1 + 64 * 256;                 // ends 135168
  float* ps2 = (float*)(ws + 135168);          // 192*256
  float* pq2 = ps2 + 192 * 256;                // ends 528384
  unsigned short* w1bf = (unsigned short*)(ws + 528384);   // 192*64      -> 552960
  unsigned short* w2bf = (unsigned short*)(ws + 552960);   // 3*48*192    -> 608256
  unsigned short* attw = (unsigned short*)(ws + 608256);   // 256*48*64   -> 2181120
  unsigned short* h1   = (unsigned short*)(ws + 2181120);  // 256*192*2048 bf16 (~201 MB)

  const float invN = 1.0f / (256.0f * 2048.0f);

  // BN1 stats + params
  k_part_f32<<<256 * 64, 256, 0, stream>>>(x, ps1, pq1, 64, 2048, 256);
  k_reduce_stats<<<64, 256, 0, stream>>>(ps1, pq1, sum1, sq1, 256);
  k_finalize<<<1, 256, 0, stream>>>(sum1, sq1, gamma1, beta1, scale1, shift1, 64, invN);
  // weight conversions
  k_cvt_w1<<<48, 256, 0, stream>>>(W1, w1bf, 192 * 64);
  k_cvt_w2<<<108, 256, 0, stream>>>(W2, w2bf);
  // conv1 (bf16 WMMA GEMM, fused BN1+ReLU)
  k_conv1<<<dim3(16, 3, 256), 256, 0, stream>>>(x, scale1, shift1, w1bf, h1);
  // BN2 stats + params
  k_part_bf16<<<256 * 192, 256, 0, stream>>>(h1, ps2, pq2, 192, 2048, 256);
  k_reduce_stats<<<192, 256, 0, stream>>>(ps2, pq2, sum2, sq2, 256);
  k_finalize<<<1, 256, 0, stream>>>(sum2, sq2, gamma2, beta2, scale2, shift2, 192, invN);
  // conv2 (3-tap shifted GEMMs, fused BN2+ReLU), h2 staged in d_out ch 64..111
  k_conv2<<<dim3(16, 1, 256), 256, 0, stream>>>(h1, scale2, shift2, w2bf, out);
  // channel attention
  k_gram<<<256, 256, 0, stream>>>(out, attw);
  k_attn<<<dim3(8, 1, 256), 256, 0, stream>>>(attw, out);
  // concat x
  k_copyx<<<4096, 256, 0, stream>>>(x, out);
}
